// Neck_12000138625192
// MI455X (gfx1250) — matrix operand order repaired and checked by reading, not compiled, not run on hardware
//
#include <hip/hip_runtime.h>
#include <hip/hip_bf16.h>

typedef __attribute__((ext_vector_type(16))) _Float16 v16h;
typedef __attribute__((ext_vector_type(8)))  _Float16 v8h;
typedef __attribute__((ext_vector_type(8)))  float    v8f;
typedef __attribute__((ext_vector_type(4)))  float    v4f;

#define EMBED 256
#define F0    512
#define F1    1024
#define MROWS 64          // rows per block (4 x 16-row M-tiles per wave)
#define XSTRIDE 264       // 256 + 8 halves pad  (528B row stride, rotates banks)
#define HSTRIDE 520       // 512 + 8 halves pad  (1040B row stride, rotates banks)

// ---------------------------------------------------------------------------
// Zero the output accumulator (harness poisons d_out; atomic-max needs 0 init).
// ---------------------------------------------------------------------------
__global__ void zero_out_kernel(float* __restrict__ out, int n) {
  int i = blockIdx.x * blockDim.x + threadIdx.x;
  if (i < n) out[i] = 0.0f;
}

// ---------------------------------------------------------------------------
// Pack f32 row-major weights W[K x N] into fragment-major f16.
// Fragment (kt, nt) covers K rows [kt*32, kt*32+32), N cols [nt*16, nt*16+16).
// B layout packed here (32x16 KxN, contiguous k order): lane = n%16 + 16*(k/16 within tile);
// element e (0..15) within lane = k%16. One thread emits one lane's 16 halves
// (32 contiguous bytes), so the GEMM kernel loads each B fragment as one v16h.
// ---------------------------------------------------------------------------
__global__ void pack_weights_kernel(const float* __restrict__ W,
                                    _Float16* __restrict__ Wp,
                                    int K, int Nn) {
  int tid = blockIdx.x * blockDim.x + threadIdx.x;
  const int NT = Nn >> 4;
  const int total = (K >> 5) * NT * 32;
  if (tid >= total) return;
  const int lane = tid & 31;
  const int tile = tid >> 5;
  const int nt   = tile % NT;
  const int kt   = tile / NT;
  const int half = (lane >> 4) & 1;
  const int n    = nt * 16 + (lane & 15);

  v8h f0, f1;
  #pragma unroll
  for (int e = 0; e < 8; ++e) {
    const int k0 = kt * 32 + 16 * half + e;
    const int k1 = k0 + 8;
    f0[e] = (_Float16)W[(size_t)k0 * Nn + n];
    f1[e] = (_Float16)W[(size_t)k1 * Nn + n];
  }
  v8h* dst = (v8h*)(Wp + (size_t)tid * 16);
  dst[0] = f0;
  dst[1] = f1;
}

// ---------------------------------------------------------------------------
// A-fragment from an f16 LDS tile. Contiguous-k A layout used here (16x32 MxK):
// lanes 0-15 hold row m=lane with K{0..15}; lanes 16-31 K{16..31}.
// Both runs are contiguous 16B -> two ds_load_b128.
// ---------------------------------------------------------------------------
__device__ __forceinline__ v16h load_afrag(const _Float16* __restrict__ base,
                                           int stride, int row, int kt, int half) {
  const _Float16* hp = base + row * stride + kt * 32 + 16 * half;
  const v8h lo = *(const v8h*)(hp);
  const v8h hi = *(const v8h*)(hp + 8);
  v16h a;
  #pragma unroll
  for (int j = 0; j < 8; ++j) { a[j] = lo[j]; a[8 + j] = hi[j]; }
  return a;
}

// ---------------------------------------------------------------------------
// Fused MLP (2x GEMM via v_wmma_f32_16x16x32_f16) + scatter-max.
// One block = one 64-row strip; 256 threads = 8 waves.
// Passes of 2 N-tiles x 4 M-tiles: each B fragment feeds 4 WMMAs, keeping
// accumulators at 64 VGPRs while halving per-row weight traffic from L2.
// ---------------------------------------------------------------------------
__global__ void __launch_bounds__(256)
mlp_scatter_kernel(const float* __restrict__ x,
                   const int*   __restrict__ gidx,
                   const _Float16* __restrict__ W1p,
                   const _Float16* __restrict__ W2p,
                   const float* __restrict__ b1,
                   const float* __restrict__ b2,
                   float* __restrict__ out,
                   int nrows) {
  __shared__ _Float16 xs[MROWS * XSTRIDE];  // 64 x 256 f16 input tile
  __shared__ _Float16 h1[MROWS * HSTRIDE];  // 64 x 512 f16 hidden tile

  const int wave = threadIdx.x >> 5;   // 0..7
  const int lane = threadIdx.x & 31;
  const int half = lane >> 4;          // A: K-subset selector / C,D: row-half
  const int lm   = lane & 15;          // A: row m / B,C,D: col n
  const int row0 = blockIdx.x * MROWS;

  // ---- Stage x tile -> LDS (f16), coalesced; clamp out-of-range rows ----
  #pragma unroll
  for (int c = 0; c < (MROWS * EMBED) / (256 * 4); ++c) {   // 16 chunks
    const int flat = (c * 256 + threadIdx.x) * 4;           // 4 floats/thread
    const int r    = flat >> 8;                             // / EMBED
    const int col  = flat & (EMBED - 1);
    int gr = row0 + r;
    gr = (gr < nrows) ? gr : (nrows - 1);
    const v4f xv = *(const v4f*)(x + (size_t)gr * EMBED + col);
    _Float16* dp = &xs[r * XSTRIDE + col];
    dp[0] = (_Float16)xv[0];
    dp[1] = (_Float16)xv[1];
    dp[2] = (_Float16)xv[2];
    dp[3] = (_Float16)xv[3];
  }
  __syncthreads();

  // Graph ids for the rows this lane-half owns in the D layout.
  int gid[4][8];
  #pragma unroll
  for (int mi = 0; mi < 4; ++mi)
    #pragma unroll
    for (int v = 0; v < 8; ++v) {
      const int r = row0 + mi * 16 + v + 8 * half;
      gid[mi][v] = (r < nrows) ? gidx[r] : -1;
    }

  // ----------------- GEMM1: xs[64x256] @ W1 -> h1[64x512] -----------------
  // Wave owns 4 N-tiles: 2 passes x 2 N-tiles x 4 M-tiles.
  for (int pass = 0; pass < 2; ++pass) {
    v8f acc[2][4];
    #pragma unroll
    for (int t = 0; t < 2; ++t) {
      const float bv = b1[(wave * 4 + pass * 2 + t) * 16 + lm];
      #pragma unroll
      for (int mi = 0; mi < 4; ++mi)
        #pragma unroll
        for (int v = 0; v < 8; ++v) acc[t][mi][v] = bv;
    }
    for (int kt = 0; kt < EMBED / 32; ++kt) {
      v16h a[4];
      #pragma unroll
      for (int mi = 0; mi < 4; ++mi)
        a[mi] = load_afrag(xs, XSTRIDE, mi * 16 + lm, kt, half);
      #pragma unroll
      for (int t = 0; t < 2; ++t) {
        const int nt = wave * 4 + pass * 2 + t;
        const v16h bfrag =
            *(const v16h*)(W1p + (((size_t)kt * (F0 / 16) + nt) * 32 + lane) * 16);
        #pragma unroll
        for (int mi = 0; mi < 4; ++mi)
          acc[t][mi] = __builtin_amdgcn_wmma_f32_16x16x32_f16(
              false, a[mi], false, bfrag, (short)0, acc[t][mi], false, false);
      }
    }
    // relu -> f16 -> LDS. D layout: elem v, lane -> (m = v + 8*half, n = lm).
    #pragma unroll
    for (int t = 0; t < 2; ++t) {
      const int n = (wave * 4 + pass * 2 + t) * 16 + lm;
      #pragma unroll
      for (int mi = 0; mi < 4; ++mi)
        #pragma unroll
        for (int v = 0; v < 8; ++v) {
          const int m = mi * 16 + v + 8 * half;
          float d = acc[t][mi][v];
          d = d > 0.0f ? d : 0.0f;
          h1[m * HSTRIDE + n] = (_Float16)d;
        }
    }
  }
  __syncthreads();

  // --------- GEMM2: h1[64x512] @ W2 -> h2[64x1024], then scatter-max -------
  // Wave owns 8 N-tiles: 4 passes x 2 N-tiles x 4 M-tiles.
  for (int pass = 0; pass < 4; ++pass) {
    v8f acc2[2][4];
    #pragma unroll
    for (int t = 0; t < 2; ++t) {
      const float bv = b2[(wave * 8 + pass * 2 + t) * 16 + lm];
      #pragma unroll
      for (int mi = 0; mi < 4; ++mi)
        #pragma unroll
        for (int v = 0; v < 8; ++v) acc2[t][mi][v] = bv;
    }
    for (int kt = 0; kt < F0 / 32; ++kt) {
      v16h a[4];
      #pragma unroll
      for (int mi = 0; mi < 4; ++mi)
        a[mi] = load_afrag(h1, HSTRIDE, mi * 16 + lm, kt, half);
      #pragma unroll
      for (int t = 0; t < 2; ++t) {
        const int nt = wave * 8 + pass * 2 + t;
        const v16h bfrag =
            *(const v16h*)(W2p + (((size_t)kt * (F1 / 16) + nt) * 32 + lane) * 16);
        #pragma unroll
        for (int mi = 0; mi < 4; ++mi)
          acc2[t][mi] = __builtin_amdgcn_wmma_f32_16x16x32_f16(
              false, a[mi], false, bfrag, (short)0, acc2[t][mi], false, false);
      }
    }
    // relu + scatter-max: non-negative floats compare as their uint bits.
    #pragma unroll
    for (int t = 0; t < 2; ++t) {
      const int n = (wave * 8 + pass * 2 + t) * 16 + lm;
      #pragma unroll
      for (int mi = 0; mi < 4; ++mi)
        #pragma unroll
        for (int v = 0; v < 8; ++v) {
          if (gid[mi][v] >= 0) {
            float d = acc2[t][mi][v];
            d = d > 0.0f ? d : 0.0f;
            atomicMax((unsigned int*)(out + (size_t)gid[mi][v] * F1 + n),
                      __float_as_uint(d));
          }
        }
    }
  }
}

// ---------------------------------------------------------------------------
extern "C" void kernel_launch(void* const* d_in, const int* in_sizes, int n_in,
                              void* d_out, int out_size, void* d_ws, size_t ws_size,
                              hipStream_t stream) {
  const float* x    = (const float*)d_in[0];  // [N, 256]
  const int*   gidx = (const int*)d_in[1];    // [N]
  // d_in[2] = num_graph (derived from out_size instead)
  const float* W1   = (const float*)d_in[3];  // [256, 512]
  const float* b1   = (const float*)d_in[4];  // [512]
  const float* W2   = (const float*)d_in[5];  // [512, 1024]
  const float* b2   = (const float*)d_in[6];  // [1024]
  float* out = (float*)d_out;                 // [G, 1024]

  const int nrows = in_sizes[0] / EMBED;

  _Float16* W1p = (_Float16*)d_ws;                 // 256*512 f16 = 256 KB
  _Float16* W2p = W1p + (size_t)EMBED * F0;        // 512*1024 f16 = 1 MB

  // 1) zero the scatter-max accumulator
  zero_out_kernel<<<(out_size + 255) / 256, 256, 0, stream>>>(out, out_size);

  // 2) pack weights to fragment-major f16
  {
    const int total1 = (EMBED / 32) * (F0 / 16) * 32;   // 8192
    pack_weights_kernel<<<(total1 + 255) / 256, 256, 0, stream>>>(W1, W1p, EMBED, F0);
    const int total2 = (F0 / 32) * (F1 / 16) * 32;      // 32768
    pack_weights_kernel<<<(total2 + 255) / 256, 256, 0, stream>>>(W2, W2p, F0, F1);
  }

  // 3) fused MLP + scatter-max, one block per 64-row strip
  const int nblocks = (nrows + MROWS - 1) / MROWS;
  mlp_scatter_kernel<<<nblocks, 256, 0, stream>>>(x, gidx, W1p, W2p, b1, b2, out, nrows);
}
